// SplatAttentionPooling_44805098832563
// MI455X (gfx1250) — matrix operand order repaired and checked by reading, not compiled, not run on hardware
//
#include <hip/hip_runtime.h>
#include <math.h>

// MI455X / gfx1250, wave32. D = A*B + C via v_wmma_f32_16x16x32_bf16.
typedef __attribute__((ext_vector_type(16))) __bf16 v16bf;
typedef __attribute__((ext_vector_type(8)))  float  v8f;
typedef __attribute__((ext_vector_type(4)))  __bf16 bf4;

#define B_DIM 64
#define C_DIM 1024
#define N_DIM 1024
#define TM 128          // rows of the Gram strip owned by one workgroup
#define TN 128          // columns per tile iteration (8 wmma tiles / wave)
#define TK 32           // k-slice per LDS stage (bf16 wmma K)
#define LDSS 40         // padded LDS row stride (elements) to avoid bank conflicts

union Frag { uint4 u[2]; v16bf v; };

// Issue the global loads for one k-slice into registers (no wait here).
__device__ __forceinline__ void load_regs(const float* __restrict__ xb, int c0, int d0,
                                          int r0, int kq, int ks,
                                          float4* ra, float4* rb)
{
    #pragma unroll
    for (int i = 0; i < 4; ++i) {
        ra[i] = *(const float4*)(xb + (size_t)(c0 + r0 + 32 * i) * N_DIM + ks + kq);
        rb[i] = *(const float4*)(xb + (size_t)(d0 + r0 + 32 * i) * N_DIM + ks + kq);
    }
}

// Convert staged fp32 registers to bf16 and store into an LDS buffer.
__device__ __forceinline__ void store_lds(__bf16* __restrict__ A, __bf16* __restrict__ Bt,
                                          int r0, int kq, const float4* ra, const float4* rb)
{
    #pragma unroll
    for (int i = 0; i < 4; ++i) {
        bf4 pa = { (__bf16)ra[i].x, (__bf16)ra[i].y, (__bf16)ra[i].z, (__bf16)ra[i].w };
        bf4 pb = { (__bf16)rb[i].x, (__bf16)rb[i].y, (__bf16)rb[i].z, (__bf16)rb[i].w };
        *(bf4*)&A [(r0 + 32 * i) * LDSS + kq] = pa;
        *(bf4*)&Bt[(r0 + 32 * i) * LDSS + kq] = pb;
    }
}

__device__ __forceinline__ void load_bfrag(const __bf16* __restrict__ Bt, int j, int l16,
                                           int kbB, Frag* f)
{
    const int brow = j * 16 + l16;
    f->u[0] = *(const uint4*)&Bt[brow * LDSS + kbB];
    f->u[1] = *(const uint4*)&Bt[brow * LDSS + kbB + 8];
}

// One workgroup: rows [c0, c0+128) of batch b's Gram matrix S = X X^T.
// Sweep 0: online row-max m and row-sumexp l (flash-softmax stats).
// Sweep 1: recompute S, accumulate column sums of exp(s-m)/l into ldsW,
// then atomically fold into global w[b, :].  Uniform scale factors (1/C mean)
// cancel later under L2 normalization.
__global__ __launch_bounds__(256)
void gram_softmax_colsum(const float* __restrict__ x, float* __restrict__ w)
{
    const int b  = blockIdx.x >> 3;            // 8 strips per batch
    const int c0 = (blockIdx.x & 7) * TM;
    const float* xb = x + (size_t)b * C_DIM * N_DIM;

    __shared__ __align__(16) __bf16 ldsA[2][TM * LDSS];
    __shared__ __align__(16) __bf16 ldsB[2][TN * LDSS];
    __shared__ float ldsW[C_DIM];

    const int t    = threadIdx.x;
    const int wave = t >> 5;
    const int lane = t & 31;
    const int l16  = lane & 15;
    const int half = lane >> 4;

    // loader geometry: thread t covers rows r0+32i, k-quad kq (4 floats)
    const int r0 = t >> 3;
    const int kq = (t & 7) << 2;

    #pragma unroll
    for (int i = 0; i < 4; ++i) ldsW[t * 4 + i] = 0.f;

    // A-fragment as used here: lane half selects K 0-15 vs 16-31 (16 contiguous; same order as B)
    const int arow = wave * 16 + l16;          // wave's 16 rows of the strip
    const int kbA  = half * 16;
    // B-fragment as used here: lane half selects K 0-15 vs 16-31 (16 contiguous; same order as A)
    const int kbB  = half * 16;

    float m_run[8], l_run[8], inv_l[8];
    #pragma unroll
    for (int r = 0; r < 8; ++r) { m_run[r] = -INFINITY; l_run[r] = 0.f; inv_l[r] = 0.f; }

    const v8f vzero = {};

    for (int sweep = 0; sweep < 2; ++sweep) {
        for (int ct = 0; ct < C_DIM / TN; ++ct) {
            const int d0 = ct * TN;
            v8f acc[8];
            #pragma unroll
            for (int j = 0; j < 8; ++j) acc[j] = vzero;

            float4 ra[4], rb[4];
            // Prologue: slice 0 -> LDS buf 0; slice 1 in-flight in registers.
            load_regs(xb, c0, d0, r0, kq, 0, ra, rb);
            store_lds(ldsA[0], ldsB[0], r0, kq, ra, rb);
            load_regs(xb, c0, d0, r0, kq, TK, ra, rb);
            __syncthreads();

            int cur = 0;
            for (int ks = 0; ks < N_DIM; ks += TK, cur ^= 1) {
                // ---- compute on LDS buffer `cur`; B-frags pipelined 1 ahead ----
                Frag a, bf[2];
                a.u[0] = *(const uint4*)&ldsA[cur][arow * LDSS + kbA];
                a.u[1] = *(const uint4*)&ldsA[cur][arow * LDSS + kbA + 8];
                load_bfrag(ldsB[cur], 0, l16, kbB, &bf[0]);
                #pragma unroll
                for (int j = 0; j < 8; ++j) {
                    if (j < 7) load_bfrag(ldsB[cur], j + 1, l16, kbB, &bf[(j + 1) & 1]);
                    acc[j] = __builtin_amdgcn_wmma_f32_16x16x32_bf16(
                        false, a.v, false, bf[j & 1].v, (short)0, acc[j], false, false);
                }
                // ---- stage next slice while WMMAs drain ----
                if (ks + TK < N_DIM) {
                    __syncthreads();   // prior reads of buf cur^1 are done
                    store_lds(ldsA[cur ^ 1], ldsB[cur ^ 1], r0, kq, ra, rb);
                    if (ks + 2 * TK < N_DIM)
                        load_regs(xb, c0, d0, r0, kq, ks + 2 * TK, ra, rb);
                    __syncthreads();   // buf cur^1 ready for next iteration
                }
            }

            // C/D layout: acc[j][r] = S[wave*16 + r + 8*half][d0 + 16j + l16]
            if (sweep == 0) {
                #pragma unroll
                for (int r = 0; r < 8; ++r) {
                    float tmax = acc[0][r];
                    #pragma unroll
                    for (int j = 1; j < 8; ++j) tmax = fmaxf(tmax, acc[j][r]);
                    #pragma unroll
                    for (int mk = 1; mk < 16; mk <<= 1)       // stays within 16-lane half
                        tmax = fmaxf(tmax, __shfl_xor(tmax, mk, 32));
                    const float mnew = fmaxf(m_run[r], tmax);
                    float tsum = 0.f;
                    #pragma unroll
                    for (int j = 0; j < 8; ++j) tsum += __expf(acc[j][r] - mnew);
                    #pragma unroll
                    for (int mk = 1; mk < 16; mk <<= 1)
                        tsum += __shfl_xor(tsum, mk, 32);
                    l_run[r] = l_run[r] * __expf(m_run[r] - mnew) + tsum;
                    m_run[r] = mnew;
                }
            } else {
                #pragma unroll
                for (int j = 0; j < 8; ++j) {
                    float cs = 0.f;
                    #pragma unroll
                    for (int r = 0; r < 8; ++r)
                        cs += __expf(acc[j][r] - m_run[r]) * inv_l[r];
                    cs += __shfl_xor(cs, 16, 32);             // fold row-halves
                    if (lane < 16)
                        atomicAdd(&ldsW[d0 + j * 16 + l16], cs);
                }
            }
            __syncthreads();   // frag reads done before next ct prologue overwrites LDS
        }
        if (sweep == 0) {
            #pragma unroll
            for (int r = 0; r < 8; ++r) inv_l[r] = 1.0f / l_run[r];  // l >= 1 (diagonal)
        }
    }

    __syncthreads();
    #pragma unroll
    for (int i = 0; i < 4; ++i)
        atomicAdd(&w[(size_t)b * C_DIM + t * 4 + i], ldsW[t * 4 + i]);
}

__global__ void zero_w(float* __restrict__ w)
{
    w[(size_t)blockIdx.x * 256 + threadIdx.x] = 0.f;
}

// p[b,n] = sum_d w[b,d] * x[b,d,n]; out = p / max(||p||, eps).
// (1/C mean factor cancels under L2 normalization.)
__global__ __launch_bounds__(1024)
void weighted_pool_normalize(const float* __restrict__ x, const float* __restrict__ w,
                             float* __restrict__ out)
{
    const int b = blockIdx.x;
    const int n = threadIdx.x;
    const float* xb = x + (size_t)b * C_DIM * N_DIM;
    const float* wb = w + (size_t)b * C_DIM;

    float acc = 0.f;
    for (int d = 0; d < C_DIM; d += 4) {
        acc += wb[d + 0] * xb[(size_t)(d + 0) * N_DIM + n];
        acc += wb[d + 1] * xb[(size_t)(d + 1) * N_DIM + n];
        acc += wb[d + 2] * xb[(size_t)(d + 2) * N_DIM + n];
        acc += wb[d + 3] * xb[(size_t)(d + 3) * N_DIM + n];
    }

    // block-wide sum of squares (32 waves of wave32)
    float sq = acc * acc;
    #pragma unroll
    for (int mk = 16; mk >= 1; mk >>= 1) sq += __shfl_xor(sq, mk, 32);
    __shared__ float red[32];
    const int wv = n >> 5, ln = n & 31;
    if (ln == 0) red[wv] = sq;
    __syncthreads();
    if (wv == 0) {
        float v = red[ln];
        #pragma unroll
        for (int mk = 16; mk >= 1; mk >>= 1) v += __shfl_xor(v, mk, 32);
        if (ln == 0) red[0] = v;
    }
    __syncthreads();
    const float norm = sqrtf(red[0]);
    out[(size_t)b * N_DIM + n] = acc / fmaxf(norm, 1e-12f);
}

extern "C" void kernel_launch(void* const* d_in, const int* in_sizes, int n_in,
                              void* d_out, int out_size, void* d_ws, size_t ws_size,
                              hipStream_t stream)
{
    const float* x  = (const float*)d_in[0];
    float* out      = (float*)d_out;
    float* w        = (float*)d_ws;          // [B, C] softmax column sums (256 KB)

    zero_w<<<B_DIM * C_DIM / 256, 256, 0, stream>>>(w);
    gram_softmax_colsum<<<B_DIM * (C_DIM / TM), 256, 0, stream>>>(x, w);
    weighted_pool_normalize<<<B_DIM, 1024, 0, stream>>>(x, w, out);
}
